// Attention_40175124087313
// MI455X (gfx1250) — compile-verified, matrix operand order checked by reading, not run on hardware
//
#include <hip/hip_runtime.h>
#include <hip/hip_bf16.h>

#define HIDDEN   2048
#define KEY_DIM  2048
#define NUM_HEADS 16
#define HEAD_DIM 128
#define SEQ      2048
#define BATCH    2
#define ROWS     (BATCH*SEQ)   // 4096

typedef __attribute__((ext_vector_type(16))) __bf16 v16bf;
typedef __attribute__((ext_vector_type(8)))  float  v8f;

union FragBF { v16bf v; unsigned short u[16]; unsigned int uu[8]; };

__device__ __forceinline__ unsigned short f2bf(float f) {
    unsigned u = __float_as_uint(f);
    u += 0x7FFFu + ((u >> 16) & 1u);      // round-to-nearest-even
    return (unsigned short)(u >> 16);
}
__device__ __forceinline__ float bf2f(unsigned short h) {
    return __uint_as_float(((unsigned)h) << 16);
}

// ---------------------------------------------------------------------------
// 1) RMSNorm: x[4096][2048] f32 -> xn bf16
// ---------------------------------------------------------------------------
__global__ void __launch_bounds__(256, 1)
rmsnorm_kernel(const float* __restrict__ x,
               const float* __restrict__ w,
               unsigned short* __restrict__ xn) {
    int row = blockIdx.x;
    const float* xr = x + (size_t)row * HIDDEN;
    float ss = 0.f;
    for (int c = threadIdx.x; c < HIDDEN; c += blockDim.x) {
        float v = xr[c];
        ss += v * v;
    }
    __shared__ float red[256];
    red[threadIdx.x] = ss;
    __syncthreads();
    for (int s = blockDim.x / 2; s > 0; s >>= 1) {
        if ((int)threadIdx.x < s) red[threadIdx.x] += red[threadIdx.x + s];
        __syncthreads();
    }
    float rstd = rsqrtf(red[0] * (1.0f / HIDDEN) + 1e-5f);
    for (int c = threadIdx.x; c < HIDDEN; c += blockDim.x)
        xn[(size_t)row * HIDDEN + c] = f2bf(xr[c] * rstd * w[c]);
}

// ---------------------------------------------------------------------------
// 2) f32 [K][N] -> bf16 transposed [N][K]  (weights, once per launch)
//    coalesced writes (consecutive threads -> consecutive k of one n-row)
// ---------------------------------------------------------------------------
__global__ void cvt_t_bf16_kernel(const float* __restrict__ src,
                                  unsigned short* __restrict__ dst,
                                  int K, int N) {
    int i = blockIdx.x * blockDim.x + threadIdx.x;   // i = n*K + k
    if (i >= K * N) return;
    int n = i / K;
    int k = i - n * K;
    dst[i] = f2bf(src[(size_t)k * N + n]);
}

// ---------------------------------------------------------------------------
// 3/6) WMMA bf16 GEMM, 64x64 per wave (4x4 WMMA tiles), BT pre-transposed [N][K]
//   MODE 0: bf16 row-major out        (+bias)
//   MODE 1: f32 out + bias + resid
//   MODE 2: bf16 out transposed as V^T [b][h][d][s]  (+bias)
// ---------------------------------------------------------------------------
template <int MODE>
__global__ void __launch_bounds__(256, 1)
gemm_wmma_kernel(const unsigned short* __restrict__ A,    // [M][K] bf16
                 const unsigned short* __restrict__ BT,   // [N][K] bf16
                 const float* __restrict__ bias,          // [N]
                 const float* __restrict__ resid,         // [M][N] (MODE 1)
                 void* __restrict__ out,
                 int M, int N, int K) {
    int wave = threadIdx.x >> 5;
    int lane = threadIdx.x & 31;
    int g    = lane >> 4;
    int m15  = lane & 15;
    int tilesN = N >> 6;                       // 64-wide macro tiles
    int tid = blockIdx.x * 8 + wave;
    int tm  = tid / tilesN;
    int tn  = tid - tm * tilesN;
    if (tm * 64 >= M) return;

    v8f acc[4][4];
#pragma unroll
    for (int i = 0; i < 4; ++i)
#pragma unroll
        for (int j = 0; j < 4; ++j) acc[i][j] = (v8f){};

    for (int k0 = 0; k0 < K; k0 += 32) {
        FragBF a[4];
#pragma unroll
        for (int i = 0; i < 4; ++i) {
            const unsigned short* ap = A + (size_t)(tm * 64 + i * 16 + m15) * K + k0;
#pragma unroll
            for (int r = 0; r < 8; ++r) {      // element pairs are K-contiguous
                int e = 2 * r;
                a[i].uu[r] = *(const unsigned int*)(ap + (e & 7) + 8 * g + 16 * (e >> 3));
            }
        }
#pragma unroll
        for (int j = 0; j < 4; ++j) {
            FragBF b;
            const unsigned short* bp = BT + (size_t)(tn * 64 + j * 16 + m15) * K + k0;
#pragma unroll
            for (int r = 0; r < 8; ++r) {
                int e = 2 * r;
                b.uu[r] = *(const unsigned int*)(bp + (e & 7) + 8 * g + 16 * (e >> 3));
            }
#pragma unroll
            for (int i = 0; i < 4; ++i)
                acc[i][j] = __builtin_amdgcn_wmma_f32_16x16x32_bf16(
                    false, a[i].v, false, b.v, (short)0, acc[i][j], false, false);
        }
    }

#pragma unroll
    for (int i = 0; i < 4; ++i) {
#pragma unroll
        for (int j = 0; j < 4; ++j) {
#pragma unroll
            for (int r = 0; r < 8; ++r) {
                int m = tm * 64 + i * 16 + r + 8 * g;     // D-frag: M = vgpr + 8*half
                int n = tn * 64 + j * 16 + m15;           // D-frag: N = lane&15
                float val = acc[i][j][r] + bias[n];
                if (MODE == 1) {
                    ((float*)out)[(size_t)m * N + n] = val + resid[(size_t)m * N + n];
                } else if (MODE == 0) {
                    ((unsigned short*)out)[(size_t)m * N + n] = f2bf(val);
                } else {                                   // V^T: [b][h][d][s]
                    int bb = m >> 11, s = m & (SEQ - 1);
                    int h  = n >> 7,  d = n & (HEAD_DIM - 1);
                    ((unsigned short*)out)[(((size_t)(bb * NUM_HEADS + h) * HEAD_DIM + d) * SEQ) + s] = f2bf(val);
                }
            }
        }
    }
}

// ---------------------------------------------------------------------------
// 4) RoPE (reference interleave semantics), bf16 -> bf16
// ---------------------------------------------------------------------------
__global__ void rope_kernel(const unsigned short* __restrict__ in,
                            unsigned short* __restrict__ outp) {
    size_t i = (size_t)blockIdx.x * blockDim.x + threadIdx.x;  // ROWS*KEY_DIM
    int col   = (int)(i & (KEY_DIM - 1));
    int row   = (int)(i >> 11);
    int s     = row & (SEQ - 1);
    int j     = col & (HEAD_DIM - 1);
    int hbase = col & ~(HEAD_DIM - 1);
    int jm    = j & 63;
    float freq = __expf(-(float)jm * (logf(10000.0f) / 64.0f));
    float ang  = (float)s * freq;
    float sn, cs;
    __sincosf(ang, &sn, &cs);
    float x1 = bf2f(in[(size_t)row * KEY_DIM + hbase + 2 * jm]);
    float x2 = bf2f(in[(size_t)row * KEY_DIM + hbase + 2 * jm + 1]);
    float r  = (j & 1) ? (sn * x1 + cs * x2) : (cs * x1 - sn * x2);
    outp[i] = f2bf(r);
}

// ---------------------------------------------------------------------------
// 5) Causal flash attention, 1 wave per (b,h,16-query tile), 32-key tiles
//    q,k: [b][s][h][d] bf16 ; v: transposed [b][h][d][s] bf16
// ---------------------------------------------------------------------------
__global__ void __launch_bounds__(32, 1)
attn_kernel(const unsigned short* __restrict__ q,
            const unsigned short* __restrict__ k,
            const unsigned short* __restrict__ vt,
            unsigned short* __restrict__ o) {
    __shared__ unsigned short pl[16 * 32];          // P tile relayout buffer
    int lane = threadIdx.x & 31;
    int bid  = blockIdx.x;                          // B*H*(S/16) = 4096
    int qt   = bid & 127;
    int h    = (bid >> 7) & 15;
    int b    = bid >> 11;
    int g    = lane >> 4;
    int m15  = lane & 15;
    int qbase = qt * 16;

    const unsigned short* qb = q  + ((size_t)(b * SEQ)) * KEY_DIM + h * HEAD_DIM;
    const unsigned short* kb = k  + ((size_t)(b * SEQ)) * KEY_DIM + h * HEAD_DIM;
    const unsigned short* vb = vt + ((size_t)(b * NUM_HEADS + h)) * HEAD_DIM * SEQ;

    // Q fragments: 4 chunks of K=32 covering HEAD_DIM=128
    FragBF qf[4];
#pragma unroll
    for (int c = 0; c < 4; ++c) {
#pragma unroll
        for (int r = 0; r < 8; ++r) {
            int e = 2 * r;
            int d = c * 32 + (e & 7) + 8 * g + 16 * (e >> 3);
            qf[c].uu[r] =
                *(const unsigned int*)(qb + (size_t)(qbase + m15) * KEY_DIM + d);
        }
    }

    v8f zero = {};
    v8f acc[8];
#pragma unroll
    for (int f = 0; f < 8; ++f) acc[f] = zero;
    float mrow[8], lrow[8];
#pragma unroll
    for (int r = 0; r < 8; ++r) { mrow[r] = -1e30f; lrow[r] = 0.f; }

    const float sc = 0.08838834764831845f;          // 1/sqrt(128)
    int ktmax = (qbase + 15) >> 5;                  // inclusive, causal
    for (int kt = 0; kt <= ktmax; ++kt) {
        // ---- scores: Q(16x128) @ K^T(128x32) -> two 16x16 frags
        v8f s0 = zero, s1 = zero;
#pragma unroll
        for (int c = 0; c < 4; ++c) {
            FragBF b0, b1;
#pragma unroll
            for (int e = 0; e < 16; ++e) {
                int d = c * 32 + (e & 7) + 8 * g + 16 * (e >> 3);
                b0.u[e] = kb[(size_t)(kt * 32 + m15) * KEY_DIM + d];
                b1.u[e] = kb[(size_t)(kt * 32 + 16 + m15) * KEY_DIM + d];
            }
            s0 = __builtin_amdgcn_wmma_f32_16x16x32_bf16(false, qf[c].v, false, b0.v,
                                                         (short)0, s0, false, false);
            s1 = __builtin_amdgcn_wmma_f32_16x16x32_bf16(false, qf[c].v, false, b1.v,
                                                         (short)0, s1, false, false);
        }
        // ---- online softmax (f32), row stats via half-wave shfl reductions
#pragma unroll
        for (int r = 0; r < 8; ++r) {
            int qrow = qbase + r + 8 * g;
            int k0i  = kt * 32 + m15;
            float a0 = (k0i      <= qrow) ? s0[r] * sc : -1e30f;
            float a1 = (k0i + 16 <= qrow) ? s1[r] * sc : -1e30f;
            float mx = fmaxf(a0, a1);
            mx = fmaxf(mx, __shfl_xor(mx, 1));
            mx = fmaxf(mx, __shfl_xor(mx, 2));
            mx = fmaxf(mx, __shfl_xor(mx, 4));
            mx = fmaxf(mx, __shfl_xor(mx, 8));
            float mnew = fmaxf(mrow[r], mx);
            float p0 = __expf(a0 - mnew);
            float p1 = __expf(a1 - mnew);
            float rs = p0 + p1;
            rs += __shfl_xor(rs, 1);
            rs += __shfl_xor(rs, 2);
            rs += __shfl_xor(rs, 4);
            rs += __shfl_xor(rs, 8);
            float corr = __expf(mrow[r] - mnew);
            lrow[r] = lrow[r] * corr + rs;
            mrow[r] = mnew;
#pragma unroll
            for (int f = 0; f < 8; ++f) acc[f][r] *= corr;
            // stage P tile (D-layout -> LDS row-major 16x32)
            pl[(r + 8 * g) * 32 + m15]      = f2bf(p0);
            pl[(r + 8 * g) * 32 + 16 + m15] = f2bf(p1);
        }
        asm volatile("s_wait_dscnt 0" ::: "memory");
        // ---- reload P as A-fragment (16x32)
        FragBF pf;
#pragma unroll
        for (int e = 0; e < 16; ++e) {
            int kk = (e & 7) + 8 * g + 16 * (e >> 3);
            pf.u[e] = pl[m15 * 32 + kk];
        }
        // ---- O += P(16x32) @ V(32x128), V^T rows contiguous in s
#pragma unroll
        for (int f = 0; f < 8; ++f) {
            FragBF bv;
#pragma unroll
            for (int r = 0; r < 8; ++r) {
                int e = 2 * r;
                int kk = (e & 7) + 8 * g + 16 * (e >> 3);
                bv.uu[r] = *(const unsigned int*)(vb + (size_t)(f * 16 + m15) * SEQ + kt * 32 + kk);
            }
            acc[f] = __builtin_amdgcn_wmma_f32_16x16x32_bf16(false, pf.v, false, bv.v,
                                                             (short)0, acc[f], false, false);
        }
        asm volatile("" ::: "memory");   // keep next iter's LDS stores after these reads
    }
    // ---- normalize + store bf16
#pragma unroll
    for (int r = 0; r < 8; ++r) {
        float inv = 1.0f / lrow[r];
        int m = qbase + r + 8 * g;
#pragma unroll
        for (int f = 0; f < 8; ++f)
            o[((size_t)(b * SEQ + m)) * KEY_DIM + h * HEAD_DIM + f * 16 + m15] =
                f2bf(acc[f][r] * inv);
    }
}

// ---------------------------------------------------------------------------
// launcher
// ---------------------------------------------------------------------------
extern "C" void kernel_launch(void* const* d_in, const int* in_sizes, int n_in,
                              void* d_out, int out_size, void* d_ws, size_t ws_size,
                              hipStream_t stream) {
    const float* x     = (const float*)d_in[0];
    const float* rms_w = (const float*)d_in[1];
    const float* wq    = (const float*)d_in[2];
    const float* bq    = (const float*)d_in[3];
    const float* wk    = (const float*)d_in[4];
    const float* bk    = (const float*)d_in[5];
    const float* wv    = (const float*)d_in[6];
    const float* bv    = (const float*)d_in[7];
    const float* wo    = (const float*)d_in[8];
    const float* bo    = (const float*)d_in[9];
    float* out = (float*)d_out;

    const size_t SZ_ACT = (size_t)ROWS * HIDDEN * 2;      // 16 MB bf16 activation
    const size_t SZ_W   = (size_t)HIDDEN * KEY_DIM * 2;   // 8 MB bf16 weight
    char* p = (char*)d_ws;
    unsigned short* xn   = (unsigned short*)p; p += SZ_ACT;
    unsigned short* wqb  = (unsigned short*)p; p += SZ_W;   // transposed [N][K]
    unsigned short* wkb  = (unsigned short*)p; p += SZ_W;
    unsigned short* wvb  = (unsigned short*)p; p += SZ_W;
    unsigned short* wob  = (unsigned short*)p; p += SZ_W;
    unsigned short* qraw = (unsigned short*)p; p += SZ_ACT;
    unsigned short* kraw = (unsigned short*)p; p += SZ_ACT;
    unsigned short* qf   = (unsigned short*)p; p += SZ_ACT;
    unsigned short* kf   = (unsigned short*)p; p += SZ_ACT;
    unsigned short* vtb  = (unsigned short*)p; p += SZ_ACT; // V^T [b][h][d][s]
    unsigned short* attn = (unsigned short*)p; p += SZ_ACT;

    // 1) RMSNorm
    rmsnorm_kernel<<<ROWS, 256, 0, stream>>>(x, rms_w, xn);

    // 2) weights -> bf16, transposed [N][K]
    int wn = HIDDEN * KEY_DIM;
    cvt_t_bf16_kernel<<<wn / 256, 256, 0, stream>>>(wq, wqb, HIDDEN, KEY_DIM);
    cvt_t_bf16_kernel<<<wn / 256, 256, 0, stream>>>(wk, wkb, HIDDEN, KEY_DIM);
    cvt_t_bf16_kernel<<<wn / 256, 256, 0, stream>>>(wv, wvb, HIDDEN, KEY_DIM);
    cvt_t_bf16_kernel<<<wn / 256, 256, 0, stream>>>(wo, wob, KEY_DIM, HIDDEN);

    // 3) QKV projections (WMMA, 64x64 per wave)
    int gemm_grid = (ROWS / 64) * (KEY_DIM / 64) / 8;     // 256
    gemm_wmma_kernel<0><<<gemm_grid, 256, 0, stream>>>(xn, wqb, bq, nullptr,
                                                       qraw, ROWS, KEY_DIM, HIDDEN);
    gemm_wmma_kernel<0><<<gemm_grid, 256, 0, stream>>>(xn, wkb, bk, nullptr,
                                                       kraw, ROWS, KEY_DIM, HIDDEN);
    gemm_wmma_kernel<2><<<gemm_grid, 256, 0, stream>>>(xn, wvb, bv, nullptr,
                                                       vtb, ROWS, KEY_DIM, HIDDEN);

    // 4) RoPE on q,k
    int rn = ROWS * KEY_DIM;
    rope_kernel<<<rn / 256, 256, 0, stream>>>(qraw, qf);
    rope_kernel<<<rn / 256, 256, 0, stream>>>(kraw, kf);

    // 5) causal attention
    attn_kernel<<<BATCH * NUM_HEADS * (SEQ / 16), 32, 0, stream>>>(qf, kf, vtb, attn);

    // 6) output projection + bias + residual (f32 out)
    gemm_wmma_kernel<1><<<gemm_grid, 256, 0, stream>>>(attn, wob, bo, x,
                                                       out, ROWS, HIDDEN, KEY_DIM);
}
